// SAE_43645457662252
// MI455X (gfx1250) — matrix operand order repaired and checked by reading, not compiled, not run on hardware
//
#include <hip/hip_runtime.h>

typedef unsigned short u16;
typedef unsigned int   u32;
typedef __attribute__((ext_vector_type(16))) __bf16 v16bf;
typedef __attribute__((ext_vector_type(8)))  float  v8f;

#define D_MODEL 2048
#define D_SAE   16384
#define NROWS   8192
#define TOPK    64

// ---------------------------------------------------------------- fp32 -> bf16 (RNE), 4-wide
__global__ void sae_cvt_bf16(const float* __restrict__ src, u16* __restrict__ dst, int n4) {
    int i = blockIdx.x * blockDim.x + threadIdx.x;
    if (i < n4) {
        const uint4 u = *(const uint4*)(src + (size_t)i * 4);
        ushort4 o;
        o.x = (u16)((u.x + 0x7FFFu + ((u.x >> 16) & 1u)) >> 16);
        o.y = (u16)((u.y + 0x7FFFu + ((u.y >> 16) & 1u)) >> 16);
        o.z = (u16)((u.z + 0x7FFFu + ((u.z >> 16) & 1u)) >> 16);
        o.w = (u16)((u.w + 0x7FFFu + ((u.w >> 16) & 1u)) >> 16);
        *(ushort4*)(dst + (size_t)i * 4) = o;
    }
}

// ---------------------------------------------------------------- WMMA fragment loads from LDS
// A fragment (16x32 bf16): lanes 0-15 = rows M0-15 holding K {0..7,16..23};
// lanes 16-31 same rows holding K {8..15,24..31}.  (ISA 7.12.2)
__device__ inline v16bf load_fragA(const u16* base, int lane, int ld) {
    int r  = lane & 15;
    int k0 = (lane >> 4) ? 8 : 0;
    const u16* p = base + r * ld + k0;
    union { v16bf v; uint4 q[2]; } u;
    u.q[0] = *(const uint4*)(p);        // K k0..k0+7
    u.q[1] = *(const uint4*)(p + 16);   // K k0+16..k0+23
    return u.v;
}
// B fragment (32x16 bf16): lanes 0-15 = cols N0-15 holding K {0..7,16..23};
// lanes 16-31 same cols holding K {8..15,24..31}.
__device__ inline v16bf load_fragB(const u16* base, int lane, int ld) {
    int c  = lane & 15;
    int k0 = (lane >> 4) ? 8 : 0;
    const u16* p = base + c * ld + k0;
    union { v16bf v; uint4 q[2]; } u;
    u.q[0] = *(const uint4*)(p);        // K k0..k0+7
    u.q[1] = *(const uint4*)(p + 16);   // K k0+16..k0+23
    return u.v;
}

// ---------------------------------------------------------------- encode GEMM: z = x * W_enc^T + b_enc
// Block: 256 threads (8 waves). Tile BM=128, BN=128, BK=64, double-buffered LDS
// fed by GLOBAL_LOAD_ASYNC_TO_LDS_B128 (ASYNCcnt).  Wave grid 4(M) x 2(N):
// each wave -> 32 rows x 64 cols = 2x4 16x16xK32 WMMA accumulators, 16 WMMA / tile.
#define BK  64
#define LDB 72   // halves per row: 64 + 8 pad (144B pitch, 16B aligned, bank-spread)

__global__ void sae_encode_wmma(const u16* __restrict__ xb, const u16* __restrict__ wb,
                                const float* __restrict__ b_enc, float* __restrict__ z) {
    __shared__ alignas(16) u16 sA[2][128 * LDB];
    __shared__ alignas(16) u16 sB[2][128 * LDB];

    const int nBase = blockIdx.x * 128;
    const int mBase = blockIdx.y * 128;
    const int tid   = threadIdx.x;
    const int lane  = tid & 31;
    const int wave  = tid >> 5;
    const int mg    = wave >> 1;   // 0..3 -> 32 rows each
    const int ng    = wave & 1;    // 0..1 -> 64 cols each

    v8f acc[2][4];
    #pragma unroll
    for (int i = 0; i < 2; ++i)
        #pragma unroll
        for (int j = 0; j < 4; ++j) acc[i][j] = (v8f)(0.0f);

    // async-issue one 128x64 bf16 tile of A and B into LDS buffer `buf`
    auto issue_tile = [&](int kb, int buf) {
        #pragma unroll
        for (int u = 0; u < 4; ++u) {
            int i = tid + u * 256;           // 1024 uint4 per matrix tile
            int row = i >> 3, seg = i & 7;   // 8 x 16B segments per 64-half row
            unsigned la = (unsigned)(unsigned long long)&sA[buf][row * LDB + seg * 8];
            const void* ga = (const void*)(xb + (size_t)(mBase + row) * D_MODEL + kb + seg * 8);
            asm volatile("global_load_async_to_lds_b128 %0, %1, off"
                         :: "v"(la), "v"(ga) : "memory");
            unsigned lb = (unsigned)(unsigned long long)&sB[buf][row * LDB + seg * 8];
            const void* gb = (const void*)(wb + (size_t)(nBase + row) * D_MODEL + kb + seg * 8);
            asm volatile("global_load_async_to_lds_b128 %0, %1, off"
                         :: "v"(lb), "v"(gb) : "memory");
        }
    };

    const int NT = D_MODEL / BK;   // 32 tiles along K
    issue_tile(0, 0);
    for (int t = 0; t < NT; ++t) {
        const int buf = t & 1;
        if (t + 1 < NT) {
            issue_tile((t + 1) * BK, buf ^ 1);
            // 8 async instructions of tile t+1 outstanding; in-order completion
            // => asynccnt<=8 means all of tile t has landed in LDS
            asm volatile("s_wait_asynccnt 0x8" ::: "memory");
        } else {
            asm volatile("s_wait_asynccnt 0x0" ::: "memory");
        }
        __syncthreads();   // every wave's share of tile t is visible

        #pragma unroll
        for (int ks = 0; ks < BK; ks += 32) {
            v16bf a[2], b[4];
            #pragma unroll
            for (int fm = 0; fm < 2; ++fm)
                a[fm] = load_fragA(&sA[buf][(mg * 32 + fm * 16) * LDB + ks], lane, LDB);
            #pragma unroll
            for (int fn = 0; fn < 4; ++fn)
                b[fn] = load_fragB(&sB[buf][(ng * 64 + fn * 16) * LDB + ks], lane, LDB);
            #pragma unroll
            for (int fm = 0; fm < 2; ++fm)
                #pragma unroll
                for (int fn = 0; fn < 4; ++fn)
                    acc[fm][fn] = __builtin_amdgcn_wmma_f32_16x16x32_bf16(
                        false, a[fm], false, b[fn], (short)0, acc[fm][fn], false, false);
        }
        __syncthreads();   // all waves done reading buf before tile t+2 overwrites it
    }

    // epilogue: C 16x16 f32 layout: lane<16 -> M=j, lane>=16 -> M=8+j; N = lane&15
    const int mo = (lane >> 4) * 8;
    const int nc = lane & 15;
    #pragma unroll
    for (int fm = 0; fm < 2; ++fm) {
        #pragma unroll
        for (int fn = 0; fn < 4; ++fn) {
            int n = nBase + ng * 64 + fn * 16 + nc;
            float be = b_enc[n];
            #pragma unroll
            for (int j = 0; j < 8; ++j) {
                int m = mBase + mg * 32 + fm * 16 + mo + j;
                z[(size_t)m * D_SAE + n] = acc[fm][fn][j] + be;
            }
        }
    }
}

// ---------------------------------------------------------------- exact per-row top-64 via radix select
__device__ inline u32 f2ord(float f) {
    u32 u = __float_as_uint(f);
    return u ^ ((u >> 31) ? 0xFFFFFFFFu : 0x80000000u);
}

__global__ void sae_topk64(const float* __restrict__ z, float* __restrict__ vals, int* __restrict__ idxo) {
    const int row = blockIdx.x;
    const float* zr = z + (size_t)row * D_SAE;
    __shared__ u32 hist[256];
    __shared__ u32 s_digit;
    __shared__ int s_remain;
    __shared__ int cA, cE;

    u32 prefix = 0, prefMask = 0;
    int remain = TOPK;

    for (int level = 0; level < 4; ++level) {
        int shift = 24 - level * 8;
        for (int i = threadIdx.x; i < 256; i += blockDim.x) hist[i] = 0;
        __syncthreads();
        for (int i = threadIdx.x; i < D_SAE; i += blockDim.x) {
            u32 k = f2ord(zr[i]);
            if ((k & prefMask) == prefix) atomicAdd(&hist[(k >> shift) & 255], 1u);
        }
        __syncthreads();
        if (threadIdx.x == 0) {
            int accu = 0, d = 255;
            for (; d >= 0; --d) {
                int c = (int)hist[d];
                if (accu + c >= remain) break;
                accu += c;
            }
            if (d < 0) d = 0;
            s_digit = (u32)d;
            s_remain = remain - accu;
        }
        __syncthreads();
        prefix   |= (s_digit << shift);
        prefMask |= (255u << shift);
        remain    = s_remain;
        __syncthreads();
    }

    // prefix == exact ordered key T of the 64th largest; remain = #ties to keep
    const u32 T = prefix;
    const int rEq = remain;
    const int baseEq = TOPK - rEq;   // count strictly greater than T
    if (threadIdx.x == 0) { cA = 0; cE = 0; }
    __syncthreads();

    float* vo = vals + row * TOPK;
    int*   io = idxo + row * TOPK;
    for (int i = threadIdx.x; i < D_SAE; i += blockDim.x) {
        float v = zr[i];
        u32 k = f2ord(v);
        float rv = v > 0.0f ? v : 0.0f;   // ReLU after selection
        if (k > T) {
            int p = atomicAdd(&cA, 1);
            vo[p] = rv; io[p] = i;
        } else if (k == T) {
            int p = atomicAdd(&cE, 1);
            if (p < rEq) { vo[baseEq + p] = rv; io[baseEq + p] = i; }
        }
    }
}

// ---------------------------------------------------------------- W_dec [2048][16384] -> Wt [16384][2048]
__global__ void sae_transpose(const float* __restrict__ W, float* __restrict__ Wt) {
    __shared__ float tile[32][33];
    const int bx = blockIdx.x;           // along D_SAE/32
    const int by = blockIdx.y;           // along D_MODEL/32
    const int tx = threadIdx.x & 31;
    const int tg = threadIdx.x >> 5;     // 8 row groups
    #pragma unroll
    for (int r = tg; r < 32; r += 8)
        tile[r][tx] = W[(size_t)(by * 32 + r) * D_SAE + bx * 32 + tx];
    __syncthreads();
    #pragma unroll
    for (int r = tg; r < 32; r += 8)
        Wt[(size_t)(bx * 32 + r) * D_MODEL + by * 32 + tx] = tile[tx][r];
}

// ---------------------------------------------------------------- sparse decode: out = feats * W_dec^T + b_dec
// Full W_dec (128 MB) < 192 MB L2; ~32 rows reuse each feature column => gathers hit L2.
__global__ void sae_decode(const float* __restrict__ vals, const int* __restrict__ idx,
                           const float* __restrict__ Wt, const float* __restrict__ b_dec,
                           float* __restrict__ out) {
    const int row = blockIdx.x;
    __shared__ float sv[TOPK];
    __shared__ int   si[TOPK];
    if (threadIdx.x < TOPK) {
        sv[threadIdx.x] = vals[row * TOPK + threadIdx.x];
        si[threadIdx.x] = idx[row * TOPK + threadIdx.x];
    }
    __syncthreads();
    for (int d = threadIdx.x; d < D_MODEL; d += blockDim.x) {
        float acc = b_dec[d];
        #pragma unroll 8
        for (int k = 0; k < TOPK; ++k) {
            float v = sv[k];
            if (v != 0.0f) acc += v * Wt[(size_t)si[k] * D_MODEL + d];
        }
        out[(size_t)row * D_MODEL + d] = acc;
    }
}

// ---------------------------------------------------------------- launch
extern "C" void kernel_launch(void* const* d_in, const int* in_sizes, int n_in,
                              void* d_out, int out_size, void* d_ws, size_t ws_size,
                              hipStream_t stream) {
    const float* x     = (const float*)d_in[0];
    // d_in[1]: position_ids (unused by the reference output)
    const float* W_enc = (const float*)d_in[2];
    const float* b_enc = (const float*)d_in[3];
    const float* W_dec = (const float*)d_in[4];
    const float* b_dec = (const float*)d_in[5];
    float* out = (float*)d_out;

    char* ws = (char*)d_ws;
    u16*   xb   = (u16*)  (ws + 0);                      //  32 MB: x as bf16
    u16*   wb   = (u16*)  (ws + (size_t)33554432);       //  64 MB: W_enc as bf16
    float* vals = (float*)(ws + (size_t)100663296);      //   2 MB
    int*   idx  = (int*)  (ws + (size_t)102760448);      //   2 MB
    float* z    = (float*)(ws + (size_t)104857600);      // 512 MB (reused below)
    float* Wt   = z;                                     // 128 MB, reused after topk

    const int nX4 = NROWS * D_MODEL / 4;   // 4,194,304 float4
    const int nW4 = D_SAE * D_MODEL / 4;   // 8,388,608 float4
    sae_cvt_bf16<<<nX4 / 256, 256, 0, stream>>>(x, xb, nX4);
    sae_cvt_bf16<<<nW4 / 256, 256, 0, stream>>>(W_enc, wb, nW4);

    sae_encode_wmma<<<dim3(D_SAE / 128, NROWS / 128), 256, 0, stream>>>(xb, wb, b_enc, z);

    sae_topk64<<<NROWS, 256, 0, stream>>>(z, vals, idx);

    sae_transpose<<<dim3(D_SAE / 32, D_MODEL / 32), 256, 0, stream>>>(W_dec, Wt);

    sae_decode<<<NROWS, 256, 0, stream>>>(vals, idx, Wt, b_dec, out);
}
